// DiffusionDecoder_43353399886320
// MI455X (gfx1250) — matrix operand order repaired and checked by reading, not compiled, not run on hardware
//
#include <hip/hip_runtime.h>
#include <hip/hip_bf16.h>

typedef __attribute__((ext_vector_type(16))) _Float16 v16h;
typedef __attribute__((ext_vector_type(8)))  _Float16 v8h;
typedef __attribute__((ext_vector_type(4)))  _Float16 v4h;
typedef __attribute__((ext_vector_type(8)))  float    v8f;

// A-matrix fragment (16x32 f16, contiguous k order): k = 16*g + s, loaded as two 8-half runs per lane.
union AFrag {
    v16h v;
    struct { v8h lo; v8h hi; } s;
};

__device__ __forceinline__ v8f wmma_f16(v16h a, v16h b, v8f c) {
    return __builtin_amdgcn_wmma_f32_16x16x32_f16(false, a, false, b, (short)0, c, false, false);
}

// ---------------------------------------------------------------------------
// fp32 -> f16 bulk convert (run once per static tensor)
// ---------------------------------------------------------------------------
__global__ __launch_bounds__(256) void cvt_f16(
    const float* __restrict__ in, _Float16* __restrict__ out, int n4)
{
    int i = blockIdx.x * 256 + threadIdx.x;
    if (i >= n4) return;
    const float4 f = ((const float4*)in)[i];
    v4h o;
    o[0] = (_Float16)f.x; o[1] = (_Float16)f.y;
    o[2] = (_Float16)f.z; o[3] = (_Float16)f.w;
    ((v4h*)out)[i] = o;
}

// ---------------------------------------------------------------------------
// GEMM: out[M,N] = X[M,K] @ W[N,K]^T + bias[N], X/W f16, acc fp32.
// Block 256 thr = 8 waves, tile 128x64, wave tile 16x64, K-chunk 32.
// EPI 0: outh = val      EPI 1: outh = val + temb[n]     EPI 2: outh = relu(val)
// EPI 3: v = xres + dt*((f32)hn + val); outf = v; outh = (f16)v
// Requires M%128==0, N%64==0, K%32==0.
// ---------------------------------------------------------------------------
template <int EPI>
__global__ __launch_bounds__(256) void gemm_h(
    const _Float16* __restrict__ X, const _Float16* __restrict__ W,
    const float* __restrict__ bias,
    _Float16* __restrict__ outh, float* __restrict__ outf,
    int M, int N, int K,
    const float* __restrict__ temb, const _Float16* __restrict__ hn,
    const float* __restrict__ xres, float dt)
{
    __shared__ __attribute__((aligned(64))) _Float16 ldsA[128][32];
    __shared__ __attribute__((aligned(64))) _Float16 ldsB[64][32];

    const int tid  = threadIdx.x;
    const int wave = tid >> 5;
    const int lane = tid & 31;
    const int g    = lane >> 4;
    const int lm   = lane & 15;
    const int m0   = blockIdx.x * 128;
    const int n0   = blockIdx.y * 64;

    v8f acc[4] = {};

    for (int k0 = 0; k0 < K; k0 += 32) {
        // stage A tile 128x32 halves: 512 16B units, pure copies
        #pragma unroll
        for (int i = 0; i < 2; ++i) {
            int u = tid + 256 * i;
            int row = u >> 2, seg = (u & 3) << 3;
            *(uint4*)&ldsA[row][seg] =
                *(const uint4*)&X[(size_t)(m0 + row) * K + k0 + seg];
        }
        // stage B tile 64x32 halves: 256 16B units
        {
            int row = tid >> 2, seg = (tid & 3) << 3;
            *(uint4*)&ldsB[row][seg] =
                *(const uint4*)&W[(size_t)(n0 + row) * K + k0 + seg];
        }
        __syncthreads();

        AFrag a;
        a.s.lo = *(const v8h*)&ldsA[wave * 16 + lm][16 * g];
        a.s.hi = *(const v8h*)&ldsA[wave * 16 + lm][16 * g + 8];
        #pragma unroll
        for (int nt = 0; nt < 4; ++nt) {
            v16h b = *(const v16h*)&ldsB[nt * 16 + lm][16 * g];
            acc[nt] = wmma_f16(a.v, b, acc[nt]);
        }
        __syncthreads();
    }

    // C/D layout: element r of lane -> m = 8*(lane>>4)+r, n = lane&15
    #pragma unroll
    for (int nt = 0; nt < 4; ++nt) {
        int n = n0 + nt * 16 + lm;
        float bn = bias[n];
        #pragma unroll
        for (int r = 0; r < 8; ++r) {
            int m = m0 + wave * 16 + 8 * g + r;
            float val = acc[nt][r] + bn;
            size_t idx = (size_t)m * N + n;
            if constexpr (EPI == 3) {
                float v2 = xres[idx] + dt * ((float)hn[idx] + val);
                outf[idx] = v2;
                outh[idx] = (_Float16)v2;
            } else {
                if constexpr (EPI == 1) val += temb[n];
                if constexpr (EPI == 2) val = fmaxf(val, 0.0f);
                outh[idx] = (_Float16)val;
            }
        }
    }
}

// ---------------------------------------------------------------------------
// Attention, one (b,h) per block. Q:64x64, K/V:256x64 (all f16), scale=1/8.
// 128 threads = 4 waves; wave w owns query rows [16w,16w+16).
// ---------------------------------------------------------------------------
__global__ __launch_bounds__(128) void attn_h(
    const _Float16* __restrict__ q, const _Float16* __restrict__ kbuf,
    const _Float16* __restrict__ vbuf, _Float16* __restrict__ ctx)
{
    __shared__ __attribute__((aligned(64))) _Float16 qs[64][64];     // 8 KB
    __shared__ __attribute__((aligned(64))) _Float16 ks[256][64];    // 32 KB (reused for P)
    __shared__ __attribute__((aligned(64))) _Float16 vts[64][256];   // 32 KB (V^T)

    const int b = blockIdx.x >> 3;
    const int h = blockIdx.x & 7;
    const int tid  = threadIdx.x;
    const int wave = tid >> 5;
    const int lane = tid & 31;
    const int g    = lane >> 4;
    const int lm   = lane & 15;

    const size_t qbase = ((size_t)b * 64) * 512 + h * 64;
    const size_t kbase = ((size_t)b * 256) * 512 + h * 64;

    // stage Q with folded 1/sqrt(64) scale (exact power of two in f16)
    #pragma unroll
    for (int i = 0; i < 4; ++i) {
        int u = tid + 128 * i;               // 512 v8h units
        int row = u >> 3, seg = (u & 7) << 3;
        v8h t = *(const v8h*)&q[qbase + (size_t)row * 512 + seg];
        t = t * (_Float16)0.125f;
        *(v8h*)&qs[row][seg] = t;
    }
    // stage K row-major: 2048 16B units, pure copies
    #pragma unroll
    for (int i = 0; i < 16; ++i) {
        int u = tid + 128 * i;
        int row = u >> 3, seg = (u & 7) << 3;
        *(uint4*)&ks[row][seg] =
            *(const uint4*)&kbuf[kbase + (size_t)row * 512 + seg];
    }
    // stage V transposed
    #pragma unroll
    for (int i = 0; i < 16; ++i) {
        int u = tid + 128 * i;
        int key = u >> 3, dseg = (u & 7) << 3;
        v8h t = *(const v8h*)&vbuf[kbase + (size_t)key * 512 + dseg];
        #pragma unroll
        for (int j = 0; j < 8; ++j) vts[dseg + j][key] = t[j];
    }
    __syncthreads();

    // S = Qs @ K^T : 16 rows x 256 keys per wave, contraction 64 (2 chunks)
    v8f sc[16] = {};
    AFrag a0, a1;
    a0.s.lo = *(const v8h*)&qs[wave * 16 + lm][16 * g];
    a0.s.hi = *(const v8h*)&qs[wave * 16 + lm][16 * g + 8];
    a1.s.lo = *(const v8h*)&qs[wave * 16 + lm][32 + 16 * g];
    a1.s.hi = *(const v8h*)&qs[wave * 16 + lm][32 + 16 * g + 8];
    #pragma unroll
    for (int nt = 0; nt < 16; ++nt) {
        v16h b0 = *(const v16h*)&ks[nt * 16 + lm][16 * g];
        v16h b1 = *(const v16h*)&ks[nt * 16 + lm][32 + 16 * g];
        sc[nt] = wmma_f16(a0.v, b0, sc[nt]);
        sc[nt] = wmma_f16(a1.v, b1, sc[nt]);
    }
    __syncthreads();   // done reading ks before overwriting with P

    // softmax: lane holds rows (8g+r), cols (16*nt+lm); reduce across 16-lane half
    float rmax[8], rsum[8];
    #pragma unroll
    for (int r = 0; r < 8; ++r) rmax[r] = -1e30f;
    #pragma unroll
    for (int nt = 0; nt < 16; ++nt)
        #pragma unroll
        for (int r = 0; r < 8; ++r) rmax[r] = fmaxf(rmax[r], sc[nt][r]);
    #pragma unroll
    for (int r = 0; r < 8; ++r) {
        #pragma unroll
        for (int m = 8; m >= 1; m >>= 1)
            rmax[r] = fmaxf(rmax[r], __shfl_xor(rmax[r], m, 32));
        rsum[r] = 0.0f;
    }
    #pragma unroll
    for (int nt = 0; nt < 16; ++nt)
        #pragma unroll
        for (int r = 0; r < 8; ++r) {
            float p = __expf(sc[nt][r] - rmax[r]);
            sc[nt][r] = p;
            rsum[r] += p;
        }
    #pragma unroll
    for (int r = 0; r < 8; ++r) {
        #pragma unroll
        for (int m = 8; m >= 1; m >>= 1)
            rsum[r] += __shfl_xor(rsum[r], m, 32);
    }

    // write unnormalized P (f16) into ks region, relaid as [64][256]
    _Float16 (*ps)[256] = reinterpret_cast<_Float16 (*)[256]>(&ks[0][0]);
    #pragma unroll
    for (int nt = 0; nt < 16; ++nt)
        #pragma unroll
        for (int r = 0; r < 8; ++r)
            ps[wave * 16 + 8 * g + r][nt * 16 + lm] = (_Float16)sc[nt][r];
    __syncthreads();

    // ctx = P @ V : 16 rows x 64 dims per wave, contraction 256 (8 chunks)
    v8f cc[4] = {};
    #pragma unroll
    for (int kc = 0; kc < 256; kc += 32) {
        AFrag pa;
        pa.s.lo = *(const v8h*)&ps[wave * 16 + lm][kc + 16 * g];
        pa.s.hi = *(const v8h*)&ps[wave * 16 + lm][kc + 16 * g + 8];
        #pragma unroll
        for (int nt = 0; nt < 4; ++nt) {
            v16h bv = *(const v16h*)&vts[nt * 16 + lm][kc + 16 * g];
            cc[nt] = wmma_f16(pa.v, bv, cc[nt]);
        }
    }

    // store f16 ctx, folding in 1/rowsum
    #pragma unroll
    for (int r = 0; r < 8; ++r) {
        float rinv = 1.0f / rsum[r];
        int l = wave * 16 + 8 * g + r;
        #pragma unroll
        for (int nt = 0; nt < 4; ++nt) {
            int d = nt * 16 + lm;
            ctx[((size_t)b * 64 + l) * 512 + h * 64 + d] =
                (_Float16)(cc[nt][r] * rinv);
        }
    }
}

// ---------------------------------------------------------------------------
// t_emb (identical across batch): temb[n] = relu(t*t1_w + t1_b) @ t2_w^T + t2_b
// ---------------------------------------------------------------------------
__global__ __launch_bounds__(256) void temb_kernel(
    const float* __restrict__ t1_w, const float* __restrict__ t1_b,
    const float* __restrict__ t2_w, const float* __restrict__ t2_b,
    float* __restrict__ temb, float tval)
{
    __shared__ float r[512];
    for (int j = threadIdx.x; j < 512; j += 256)
        r[j] = fmaxf(tval * t1_w[j] + t1_b[j], 0.0f);
    __syncthreads();
    for (int n = threadIdx.x; n < 512; n += 256) {
        float s = t2_b[n];
        const float* w = &t2_w[(size_t)n * 512];
        for (int j = 0; j < 512; ++j) s = fmaf(r[j], w[j], s);
        temb[n] = s;
    }
}

// ---------------------------------------------------------------------------
// LayerNorm over A=128: hn = LN(x + vt)*g + b. x fp32, vt f16, hn f16.
// One wave per row.
// ---------------------------------------------------------------------------
__global__ __launch_bounds__(256) void ln_h(
    const float* __restrict__ x, const _Float16* __restrict__ vt,
    const float* __restrict__ gam, const float* __restrict__ bet,
    _Float16* __restrict__ hn)
{
    const int row  = blockIdx.x * 8 + (threadIdx.x >> 5);
    const int lane = threadIdx.x & 31;
    const float4 xr = ((const float4*)(x + (size_t)row * 128))[lane];
    const v4h vv = *(const v4h*)&vt[(size_t)row * 128 + lane * 4];
    float h0 = xr.x + (float)vv[0], h1 = xr.y + (float)vv[1];
    float h2 = xr.z + (float)vv[2], h3 = xr.w + (float)vv[3];
    float s  = h0 + h1 + h2 + h3;
    float ss = h0*h0 + h1*h1 + h2*h2 + h3*h3;
    #pragma unroll
    for (int m = 16; m >= 1; m >>= 1) {
        s  += __shfl_xor(s,  m, 32);
        ss += __shfl_xor(ss, m, 32);
    }
    const float mu  = s * (1.0f / 128.0f);
    const float var = ss * (1.0f / 128.0f) - mu * mu;
    const float rs  = rsqrtf(var + 1e-5f);
    const float4 g4 = ((const float4*)gam)[lane];
    const float4 b4 = ((const float4*)bet)[lane];
    v4h o;
    o[0] = (_Float16)((h0 - mu) * rs * g4.x + b4.x);
    o[1] = (_Float16)((h1 - mu) * rs * g4.y + b4.y);
    o[2] = (_Float16)((h2 - mu) * rs * g4.z + b4.z);
    o[3] = (_Float16)((h3 - mu) * rs * g4.w + b4.w);
    *(v4h*)&hn[(size_t)row * 128 + lane * 4] = o;
}

// ---------------------------------------------------------------------------
extern "C" void kernel_launch(void* const* d_in, const int* in_sizes, int n_in,
                              void* d_out, int out_size, void* d_ws, size_t ws_size,
                              hipStream_t stream)
{
    const float* cond   = (const float*)d_in[0];
    const float* noise  = (const float*)d_in[1];
    const float* t1_w   = (const float*)d_in[2];
    const float* t1_b   = (const float*)d_in[3];
    const float* t2_w   = (const float*)d_in[4];
    const float* t2_b   = (const float*)d_in[5];
    const float* qp_w   = (const float*)d_in[6];
    const float* qp_b   = (const float*)d_in[7];
    const float* in_w   = (const float*)d_in[8];
    const float* in_b   = (const float*)d_in[9];
    const float* op_w   = (const float*)d_in[10];
    const float* op_b   = (const float*)d_in[11];
    const float* outp_w = (const float*)d_in[12];
    const float* outp_b = (const float*)d_in[13];
    const float* f1_w   = (const float*)d_in[14];
    const float* f1_b   = (const float*)d_in[15];
    const float* f2_w   = (const float*)d_in[16];
    const float* f2_b   = (const float*)d_in[17];
    const float* ln_g   = (const float*)d_in[18];
    const float* ln_b   = (const float*)d_in[19];

    // workspace carve: fp32 region then f16 region
    float* wsf = (float*)d_ws;
    float* temb = wsf; wsf += 512;
    float* xA   = wsf; wsf += (size_t)8192 * 128;
    float* xB   = wsf; wsf += (size_t)8192 * 128;
    _Float16* wsh = (_Float16*)wsf;
    _Float16* condh  = wsh; wsh += (size_t)32768 * 512;
    _Float16* kh     = wsh; wsh += (size_t)32768 * 512;
    _Float16* vh     = wsh; wsh += (size_t)32768 * 512;
    _Float16* winh   = wsh; wsh += (size_t)1536 * 512;
    _Float16* wqph   = wsh; wsh += (size_t)512 * 128;
    _Float16* woph   = wsh; wsh += (size_t)512 * 512;
    _Float16* woutph = wsh; wsh += (size_t)128 * 512;
    _Float16* wf1h   = wsh; wsh += (size_t)512 * 128;
    _Float16* wf2h   = wsh; wsh += (size_t)128 * 512;
    _Float16* qinh   = wsh; wsh += (size_t)8192 * 512;   // reused as ctx
    _Float16* qh     = wsh; wsh += (size_t)8192 * 512;
    _Float16* obufh  = wsh; wsh += (size_t)8192 * 512;   // reused as ffn hidden
    _Float16* vtbh   = wsh; wsh += (size_t)8192 * 128;
    _Float16* hnh    = wsh; wsh += (size_t)8192 * 128;
    _Float16* xh0    = wsh; wsh += (size_t)8192 * 128;
    _Float16* xhA    = wsh; wsh += (size_t)8192 * 128;
    _Float16* xhB    = wsh; wsh += (size_t)8192 * 128;
    _Float16* ctxh = qinh;
    _Float16* fbuf = obufh;

    auto cvt = [&](const float* in, _Float16* out, size_t n) {
        int n4 = (int)(n / 4);
        cvt_f16<<<dim3((n4 + 255) / 256), dim3(256), 0, stream>>>(in, out, n4);
    };
    // one-time conversions of static tensors + initial state
    cvt(cond,   condh,  (size_t)32768 * 512);
    cvt(in_w,   winh,   (size_t)1536 * 512);
    cvt(qp_w,   wqph,   (size_t)512 * 128);
    cvt(op_w,   woph,   (size_t)512 * 512);
    cvt(outp_w, woutph, (size_t)128 * 512);
    cvt(f1_w,   wf1h,   (size_t)512 * 128);
    cvt(f2_w,   wf2h,   (size_t)128 * 512);
    cvt(noise,  xh0,    (size_t)8192 * 128);

    const dim3 blk(256);

    // K/V projections (once)
    gemm_h<0><<<dim3(256, 8), blk, 0, stream>>>(
        condh, winh + (size_t)512 * 512, in_b + 512, kh, nullptr,
        32768, 512, 512, nullptr, nullptr, nullptr, 0.0f);
    gemm_h<0><<<dim3(256, 8), blk, 0, stream>>>(
        condh, winh + (size_t)2 * 512 * 512, in_b + 2 * 512, vh, nullptr,
        32768, 512, 512, nullptr, nullptr, nullptr, 0.0f);

    const float dt = -1.0f / 20.0f;
    const float* xin = noise;
    const _Float16* xinh = xh0;
    for (int s = 0; s < 20; ++s) {
        const float tv = 1.0f + dt * (float)s;
        float*     xout  = (s == 19) ? (float*)d_out : ((s & 1) ? xB : xA);
        _Float16*  xhout = (s & 1) ? xhB : xhA;

        temb_kernel<<<1, 256, 0, stream>>>(t1_w, t1_b, t2_w, t2_b, temb, tv);
        // q_in = x @ qp_w^T + qp_b + temb
        gemm_h<1><<<dim3(64, 8), blk, 0, stream>>>(
            xinh, wqph, qp_b, qinh, nullptr, 8192, 512, 128,
            temb, nullptr, nullptr, 0.0f);
        // q = q_in @ in_w[:E]^T + in_b
        gemm_h<0><<<dim3(64, 8), blk, 0, stream>>>(
            qinh, winh, in_b, qh, nullptr, 8192, 512, 512,
            nullptr, nullptr, nullptr, 0.0f);
        // attention -> ctx (overwrites qinh)
        attn_h<<<dim3(1024), dim3(128), 0, stream>>>(qh, kh, vh, ctxh);
        // ctx @ op_w^T + op_b
        gemm_h<0><<<dim3(64, 8), blk, 0, stream>>>(
            ctxh, woph, op_b, obufh, nullptr, 8192, 512, 512,
            nullptr, nullptr, nullptr, 0.0f);
        // v_t = . @ outp_w^T + outp_b
        gemm_h<0><<<dim3(64, 2), blk, 0, stream>>>(
            obufh, woutph, outp_b, vtbh, nullptr, 8192, 128, 512,
            nullptr, nullptr, nullptr, 0.0f);
        // hn = LN(x + v_t)
        ln_h<<<dim3(1024), dim3(256), 0, stream>>>(xin, vtbh, ln_g, ln_b, hnh);
        // hidden = relu(hn @ f1_w^T + f1_b)   (overwrites obufh)
        gemm_h<2><<<dim3(64, 8), blk, 0, stream>>>(
            hnh, wf1h, f1_b, fbuf, nullptr, 8192, 512, 128,
            nullptr, nullptr, nullptr, 0.0f);
        // x_out = x + dt*(hn + hidden @ f2_w^T + f2_b), dual fp32/f16 store
        gemm_h<3><<<dim3(64, 2), blk, 0, stream>>>(
            fbuf, wf2h, f2_b, xhout, xout, 8192, 128, 512,
            nullptr, hnh, xin, dt);

        xin = xout;
        xinh = xhout;
    }
}
